// DilatedAttention_63995012711033
// MI455X (gfx1250) — compile-verified, matrix operand order checked by reading, not run on hardware
//
#include <hip/hip_runtime.h>

// ---------------------------------------------------------------------------
// LongNet dilated attention: B=2, T=4096, C=1024, H=16, D=64
// SEGMENTS=((512,1),(1024,2),(2048,4)); eff = w/r = 512 for every branch.
// ---------------------------------------------------------------------------

typedef _Float16 half_t;
typedef __attribute__((ext_vector_type(16))) _Float16     v16h;
typedef __attribute__((ext_vector_type(8)))  float        v8f;
typedef __attribute__((ext_vector_type(4)))  unsigned int u32x4;
typedef __attribute__((ext_vector_type(8)))  int          i32x8;
typedef __attribute__((ext_vector_type(4)))  int          i32x4;

#if defined(__has_builtin)
#if __has_builtin(__builtin_amdgcn_tensor_load_to_lds) && \
    __has_builtin(__builtin_amdgcn_s_wait_tensorcnt)
#define HAVE_TDM 1
#endif
#endif
#ifndef HAVE_TDM
#define HAVE_TDM 0
#endif

union FragU { v16h v; u32x4 q[2]; };
union H8U   { u32x4 q; half_t h[8]; };

// Build one 16-lane-striped f16 operand fragment (A 16x32 or B 32x16) from a
// row-major run: elements 0..7 at p[0..7], elements 8..15 at p[16..23].
// Caller supplies p already offset by 8*(lane>>4) in the K direction per the
// CDNA5 ISA 7.12.2 layout.
static __device__ __forceinline__ v16h load_frag(const half_t* p) {
  FragU f;
  f.q[0] = *(const u32x4*)p;
  f.q[1] = *(const u32x4*)(p + 16);
  return f.v;
}

static __device__ __forceinline__ v8f wmma_f16(v16h a, v16h b, v8f c) {
  // v_wmma_f32_16x16x32_f16  D = A*B + C, f32 accumulate
  return __builtin_amdgcn_wmma_f32_16x16x32_f16(false, a, false, b,
                                                (short)0, c, false, false);
}

// ---------------------------------------------------------------------------
// Elementwise prep kernels
// ---------------------------------------------------------------------------
__global__ __launch_bounds__(256) void conv_f16_kernel(const float* __restrict__ in,
                                                       half_t* __restrict__ out,
                                                       long n) {
  long stride = (long)gridDim.x * blockDim.x;
  for (long i = (long)blockIdx.x * blockDim.x + threadIdx.x; i < n; i += stride)
    out[i] = (half_t)in[i];
}

// Wt[n*K + k] = (half)W[k*N + n]   (transpose + downconvert)
__global__ __launch_bounds__(256) void conv_transpose_kernel(const float* __restrict__ W,
                                                             half_t* __restrict__ Wt,
                                                             int K, int N) {
  long total = (long)K * N;
  long stride = (long)gridDim.x * blockDim.x;
  for (long i = (long)blockIdx.x * blockDim.x + threadIdx.x; i < total; i += stride) {
    long n = i / K, k = i - n * K;
    Wt[i] = (half_t)W[k * (long)N + n];
  }
}

__global__ __launch_bounds__(256) void fill_f32_kernel(float* __restrict__ p, float v, long n) {
  long stride = (long)gridDim.x * blockDim.x;
  for (long i = (long)blockIdx.x * blockDim.x + threadIdx.x; i < n; i += stride)
    p[i] = v;
}

// ---------------------------------------------------------------------------
// WMMA GEMM:  C[M,N] = A[M,K] @ Bt[N,K]^T + bias[N]
//   block = 256 threads (8 waves), tile 128x128, k-step 32
//   wave (wm in 0..3, wn in 0..1) computes a 32x64 sub-tile = 2x4 WMMA tiles
// OUT_MODE 0: f16 row-major (ldc)
// OUT_MODE 1: f16 scattered to [B,H,T,D] with post-bias scale (q/k/v proj)
// OUT_MODE 2: f32 row-major (ldc)  -> final output
// ---------------------------------------------------------------------------
template <int OUT_MODE>
__global__ __launch_bounds__(256) void gemm_wmma_kernel(
    const half_t* __restrict__ A, int lda,
    const half_t* __restrict__ Bt,
    const float* __restrict__ bias,
    void* __restrict__ Cout, int ldc,
    int K, float scale) {
  __shared__ half_t As[128 * 40];
  __shared__ half_t Bs[128 * 40];

  const int tid = threadIdx.x;
  const int m0 = blockIdx.y * 128;
  const int n0 = blockIdx.x * 128;
  const int wave = tid >> 5;
  const int lane = tid & 31;
  const int wm = wave & 3;        // 4 waves along M
  const int wn = wave >> 2;       // 2 waves along N
  const int half_sel = lane >> 4; // which 16-lane half
  const int rlo = lane & 15;
  const int kb8 = half_sel * 8;   // K phase offset per ISA fragment layout

  v8f acc[2][4];
#pragma unroll
  for (int mt = 0; mt < 2; mt++)
#pragma unroll
    for (int nt = 0; nt < 4; nt++)
      acc[mt][nt] = (v8f){0.f, 0.f, 0.f, 0.f, 0.f, 0.f, 0.f, 0.f};

  const int srow = tid >> 2;        // 0..63
  const int sc8 = (tid & 3) * 8;    // 0,8,16,24

  for (int kb = 0; kb < K; kb += 32) {
#pragma unroll
    for (int p = 0; p < 2; p++) {
      int rr = srow + p * 64;
      *(u32x4*)&As[rr * 40 + sc8] =
          *(const u32x4*)&A[(size_t)(m0 + rr) * lda + kb + sc8];
      *(u32x4*)&Bs[rr * 40 + sc8] =
          *(const u32x4*)&Bt[(size_t)(n0 + rr) * K + kb + sc8];
    }
    __syncthreads();

    v16h af[2], bf[4];
#pragma unroll
    for (int mt = 0; mt < 2; mt++)
      af[mt] = load_frag(&As[(wm * 32 + mt * 16 + rlo) * 40 + kb8]);
#pragma unroll
    for (int nt = 0; nt < 4; nt++)
      bf[nt] = load_frag(&Bs[(wn * 64 + nt * 16 + rlo) * 40 + kb8]);

#pragma unroll
    for (int mt = 0; mt < 2; mt++)
#pragma unroll
      for (int nt = 0; nt < 4; nt++)
        acc[mt][nt] = wmma_f16(af[mt], bf[nt], acc[mt][nt]);

    __syncthreads();
  }

  // Epilogue: C fragment layout (ISA 7.12.2): vgpr i, lanes 0-15 -> row i,
  // lanes 16-31 -> row i+8; column = lane&15.
#pragma unroll
  for (int mt = 0; mt < 2; mt++) {
#pragma unroll
    for (int nt = 0; nt < 4; nt++) {
#pragma unroll
      for (int i = 0; i < 8; i++) {
        int row = m0 + wm * 32 + mt * 16 + i + 8 * half_sel;
        int col = n0 + wn * 64 + nt * 16 + rlo;
        float v = acc[mt][nt][i] + bias[col];
        if (OUT_MODE == 0) {
          ((half_t*)Cout)[(size_t)row * ldc + col] = (half_t)v;
        } else if (OUT_MODE == 1) {
          v *= scale;
          int b = row >> 12;      // T = 4096
          int t = row & 4095;
          int h = col >> 6;       // D = 64
          int d = col & 63;
          size_t dst = (((size_t)(b * 16 + h) * 4096 + t) << 6) | d;
          ((half_t*)Cout)[dst] = (half_t)v;
        } else {
          ((float*)Cout)[(size_t)row * ldc + col] = v;
        }
      }
    }
  }
}

// ---------------------------------------------------------------------------
// Dilated attention branch.
// grid = B*H*nseg blocks, 512 threads (16 waves). Each block handles one
// (b, h, segment): eff = 512 gathered positions, causal, D = 64.
// Each wave owns 32 query rows [wave*32, wave*32+32).
// Dynamic LDS: Ks[512][64] f16 | Vt[64][512] f16 | Pw[16][32][32] f16
// The dilated K gather (row stride = r*64 elements) is expressed as a single
// Tensor Data Mover descriptor: tile 64x512, tensor_dim0_stride = r*64.
// ---------------------------------------------------------------------------
__global__ __launch_bounds__(512) void attn_branch_kernel(
    const half_t* __restrict__ Qg, const half_t* __restrict__ Kg,
    const half_t* __restrict__ Vg,
    float* __restrict__ Obuf, float* __restrict__ Lbuf,
    int w, int r, int nseg) {
  extern __shared__ half_t smem[];
  half_t* Ks = smem;                 // 512*64 (row-major)
  half_t* Vt = smem + 512 * 64;      // 64*512 (transposed)
  half_t* Pb = smem + 2 * 512 * 64;  // 16 * 32*32

  const int tid = threadIdx.x;
  const int s = blockIdx.x % nseg;
  const int bh = blockIdx.x / nseg;         // b*16 + h
  const int h = bh & 15;
  const int off = h % r;                    // LongNet head offset
  const size_t baseRow = (size_t)bh * 4096; // rows in [B*H, T] space
  const int segBase = s * w + off;

#if HAVE_TDM
  // ---- TDM: stage the dilated K tile (512 rows x 64 halfs, row stride r*64)
  if (tid < 32) {
    unsigned long long ga =
        (unsigned long long)(const void*)(Kg + (baseRow + segBase) * 64);
    unsigned int lds_off = (unsigned int)(unsigned long long)(void*)Ks;
    u32x4 g0;
    g0[0] = 1u;                                   // count=1, user descriptor
    g0[1] = lds_off;                              // lds_addr
    g0[2] = (unsigned int)ga;                     // global_addr[31:0]
    g0[3] = (unsigned int)((ga >> 32) & 0x1FFFFFFull) | (2u << 30); // type=2
    unsigned long long stride0 = (unsigned long long)r * 64;        // elements
    i32x8 g1;
    g1[0] = 0x00010000;                           // data_size=1 (2 bytes)
    g1[1] = (int)(64u << 16);                     // tensor_dim0 = 64 (lo16<<16)
    g1[2] = (int)(512u << 16);                    // dim0 hi | tensor_dim1=512
    g1[3] = (int)(64u << 16);                     // dim1 hi | tile_dim0=64
    g1[4] = 512;                                  // tile_dim1=512, tile_dim2=0
    g1[5] = (int)(unsigned int)stride0;           // tensor_dim0_stride lo
    g1[6] = (int)(unsigned int)(stride0 >> 32);   // stride0 hi | stride1 lo16
    g1[7] = 0;                                    // tensor_dim1_stride hi
    i32x4 gz4 = (i32x4){0, 0, 0, 0};
    i32x8 gz8 = (i32x8){0, 0, 0, 0, 0, 0, 0, 0};
    __builtin_amdgcn_tensor_load_to_lds(g0, g1, gz4, gz4, gz8, 0);
    __builtin_amdgcn_s_wait_tensorcnt((short)0);
  }
#endif

  // ---- stage V transposed into LDS (and K manually if no TDM)
  {
    int j = tid; // gathered key index 0..511
    int pos = segBase + j * r;
#if !HAVE_TDM
    const half_t* krow = Kg + (baseRow + pos) * 64;
#pragma unroll
    for (int c = 0; c < 8; c++)
      ((u32x4*)(Ks + (size_t)j * 64))[c] = ((const u32x4*)krow)[c];
#endif
    const half_t* vrow = Vg + (baseRow + pos) * 64;
#pragma unroll
    for (int c = 0; c < 8; c++) {
      H8U u;
      u.q = ((const u32x4*)vrow)[c];
#pragma unroll
      for (int e = 0; e < 8; e++)
        Vt[(size_t)(c * 8 + e) * 512 + j] = u.h[e];
    }
  }
  __syncthreads();

  const int wave = tid >> 5;
  const int lane = tid & 31;
  const int half_sel = lane >> 4;
  const int rlo = lane & 15;
  const int kb8 = half_sel * 8;
  const int qbase = wave * 32;

  // ---- load Q fragments straight from global (gathered rows, 2 qtiles x 2 k)
  v16h qf[2][2];
#pragma unroll
  for (int mt = 0; mt < 2; mt++) {
    int row = qbase + mt * 16 + rlo;
    int pos = segBase + row * r;
    const half_t* qrow = Qg + (baseRow + pos) * 64;
#pragma unroll
    for (int kt = 0; kt < 2; kt++)
      qf[mt][kt] = load_frag(qrow + kt * 32 + kb8);
  }

  v8f oacc[2][4];
#pragma unroll
  for (int mt = 0; mt < 2; mt++)
#pragma unroll
    for (int dt = 0; dt < 4; dt++)
      oacc[mt][dt] = (v8f){0.f, 0.f, 0.f, 0.f, 0.f, 0.f, 0.f, 0.f};

  float mrun[2][8], lrun[2][8];
#pragma unroll
  for (int mt = 0; mt < 2; mt++)
#pragma unroll
    for (int i = 0; i < 8; i++) { mrun[mt][i] = -1e30f; lrun[mt][i] = 0.f; }

  half_t* Pw = Pb + (size_t)wave * (32 * 32);
  const int ntiles = (qbase + 31) / 32 + 1; // causal: key tiles of 32

  for (int jt = 0; jt < ntiles; jt++) {
    const int j0 = jt * 32;

    // ---- S = Q K^T  (f32 accum), 2 qtiles x 2 key-halves, K-dim = 64
    v8f sacc[2][2];
#pragma unroll
    for (int mt = 0; mt < 2; mt++) {
#pragma unroll
      for (int nt = 0; nt < 2; nt++) {
        v8f a = (v8f){0.f, 0.f, 0.f, 0.f, 0.f, 0.f, 0.f, 0.f};
#pragma unroll
        for (int kt = 0; kt < 2; kt++) {
          const half_t* bp =
              Ks + (size_t)(j0 + nt * 16 + rlo) * 64 + kt * 32 + kb8;
          a = wmma_f16(qf[mt][kt], load_frag(bp), a);
        }
        sacc[mt][nt] = a;
      }
    }

    // ---- online softmax (rows live in 16-lane halves; butterfly reduce)
#pragma unroll
    for (int mt = 0; mt < 2; mt++) {
      float e0v[8], e1v[8];
#pragma unroll
      for (int i = 0; i < 8; i++) {
        int row = qbase + mt * 16 + i + 8 * half_sel;
        float s0 = sacc[mt][0][i];
        float s1 = sacc[mt][1][i];
        if (j0 + rlo > row) s0 = -1e30f;
        if (j0 + 16 + rlo > row) s1 = -1e30f;
        float mx = fmaxf(s0, s1);
#pragma unroll
        for (int o = 8; o >= 1; o >>= 1)
          mx = fmaxf(mx, __shfl_xor(mx, o, 32));
        float mnew = fmaxf(mrun[mt][i], mx);
        float scl = __expf(mrun[mt][i] - mnew);
        float e0 = __expf(s0 - mnew);
        float e1 = __expf(s1 - mnew);
        float rs = e0 + e1;
#pragma unroll
        for (int o = 8; o >= 1; o >>= 1)
          rs += __shfl_xor(rs, o, 32);
        lrun[mt][i] = lrun[mt][i] * scl + rs;
        mrun[mt][i] = mnew;
#pragma unroll
        for (int dt = 0; dt < 4; dt++)
          oacc[mt][dt][i] *= scl;
        e0v[i] = e0;
        e1v[i] = e1;
      }
      // stage P (f16) into the per-wave LDS scratch, row-major stride 32
#pragma unroll
      for (int i = 0; i < 8; i++) {
        int rl = mt * 16 + i + 8 * half_sel;
        Pw[rl * 32 + rlo] = (half_t)e0v[i];
        Pw[rl * 32 + 16 + rlo] = (half_t)e1v[i];
      }
    }
    // per-wave DS ops are issued & retired in order; wait so every lane's P
    // stores are visible before the cross-lane fragment gather below
    asm volatile("s_wait_dscnt 0x0" ::: "memory");

    // ---- O += P V  (A = P 16x32 from LDS, B = Vt rows, contiguous)
#pragma unroll
    for (int mt = 0; mt < 2; mt++) {
      v16h pf = load_frag(Pw + (mt * 16 + rlo) * 32 + kb8);
#pragma unroll
      for (int dt = 0; dt < 4; dt++) {
        v16h vf = load_frag(Vt + (size_t)(dt * 16 + rlo) * 512 + j0 + kb8);
        oacc[mt][dt] = wmma_f16(pf, vf, oacc[mt][dt]);
      }
    }
  }

  // ---- epilogue: normalize, scatter O and lse back to full positions
#pragma unroll
  for (int mt = 0; mt < 2; mt++) {
#pragma unroll
    for (int i = 0; i < 8; i++) {
      int rowl = qbase + mt * 16 + i + 8 * half_sel;
      int pos = segBase + rowl * r;
      float inv = 1.0f / lrun[mt][i];
      float* orow = Obuf + (baseRow + pos) * 64;
#pragma unroll
      for (int dt = 0; dt < 4; dt++)
        orow[dt * 16 + rlo] = oacc[mt][dt][i] * inv;
      if (rlo == 0)
        Lbuf[baseRow + pos] = mrun[mt][i] + __logf(lrun[mt][i]);
    }
  }
}

// ---------------------------------------------------------------------------
// LSE-weighted combine of the 3 branches -> y[B,T,C] in f16
// ---------------------------------------------------------------------------
__global__ __launch_bounds__(256) void combine_kernel(
    const float* __restrict__ O0, const float* __restrict__ O1,
    const float* __restrict__ O2, const float* __restrict__ L0,
    const float* __restrict__ L1, const float* __restrict__ L2,
    half_t* __restrict__ Y) {
  int idx = blockIdx.x * blockDim.x + threadIdx.x; // over B*H*T = 131072
  int t = idx & 4095;
  int bh = idx >> 12;
  int h = bh & 15;
  int b = bh >> 4;
  float l0 = L0[idx], l1 = L1[idx], l2 = L2[idx];
  float m = fmaxf(l0, fmaxf(l1, l2));
  float w0 = __expf(l0 - m), w1 = __expf(l1 - m), w2 = __expf(l2 - m);
  float inv = 1.0f / (w0 + w1 + w2);
  const float* o0 = O0 + (size_t)idx * 64;
  const float* o1 = O1 + (size_t)idx * 64;
  const float* o2 = O2 + (size_t)idx * 64;
  half_t* y = Y + ((size_t)(b * 4096 + t)) * 1024 + h * 64;
#pragma unroll
  for (int d8 = 0; d8 < 8; d8++) {
    H8U u;
#pragma unroll
    for (int e = 0; e < 8; e++) {
      int d = d8 * 8 + e;
      u.h[e] = (half_t)((w0 * o0[d] + w1 * o1[d] + w2 * o2[d]) * inv);
    }
    *(u32x4*)(y + d8 * 8) = u.q;
  }
}

// ---------------------------------------------------------------------------
// Host launch
// ---------------------------------------------------------------------------
extern "C" void kernel_launch(void* const* d_in, const int* in_sizes, int n_in,
                              void* d_out, int out_size, void* d_ws, size_t ws_size,
                              hipStream_t stream) {
  const float* x        = (const float*)d_in[0];
  const float* c_attn_w = (const float*)d_in[1];
  const float* c_attn_b = (const float*)d_in[2];
  const float* q_w      = (const float*)d_in[3];
  const float* q_b      = (const float*)d_in[4];
  const float* k_w      = (const float*)d_in[5];
  const float* k_b      = (const float*)d_in[6];
  const float* v_w      = (const float*)d_in[7];
  const float* v_b      = (const float*)d_in[8];
  const float* o_w      = (const float*)d_in[9];
  const float* o_b      = (const float*)d_in[10];
  float* out = (float*)d_out;

  // workspace carve-out
  size_t offset = 0;
  auto alloc = [&](size_t bytes) -> void* {
    void* p = (char*)d_ws + offset;
    offset = (offset + bytes + 255) & ~(size_t)255;
    return p;
  };
  const size_t MT = 8192;        // B*T
  half_t* xh     = (half_t*)alloc(MT * 1024 * 2);
  half_t* wAttnT = (half_t*)alloc(3072ull * 1024 * 2);
  half_t* qwT    = (half_t*)alloc(1024ull * 1024 * 2);
  half_t* kwT    = (half_t*)alloc(1024ull * 1024 * 2);
  half_t* vwT    = (half_t*)alloc(1024ull * 1024 * 2);
  half_t* owT    = (half_t*)alloc(1024ull * 1024 * 2);
  half_t* qkvh   = (half_t*)alloc(MT * 3072 * 2);
  half_t* qbuf   = (half_t*)alloc(8388608ull * 2); // [B,H,T,D] f16
  half_t* kbuf   = (half_t*)alloc(8388608ull * 2);
  half_t* vbuf   = (half_t*)alloc(8388608ull * 2);
  float*  O0     = (float*)alloc(8388608ull * 4);
  float*  O1     = (float*)alloc(8388608ull * 4);
  float*  O2     = (float*)alloc(8388608ull * 4);
  float*  L0     = (float*)alloc(131072ull * 4);
  float*  L1     = (float*)alloc(131072ull * 4);
  float*  L2     = (float*)alloc(131072ull * 4);
  half_t* yh     = (half_t*)alloc(MT * 1024 * 2);
  (void)ws_size; (void)in_sizes; (void)n_in; (void)out_size;

  // 1) downconvert x, transpose+downconvert weights
  conv_f16_kernel<<<4096, 256, 0, stream>>>(x, xh, (long)MT * 1024);
  conv_transpose_kernel<<<2048, 256, 0, stream>>>(c_attn_w, wAttnT, 1024, 3072);
  conv_transpose_kernel<<<1024, 256, 0, stream>>>(q_w, qwT, 1024, 1024);
  conv_transpose_kernel<<<1024, 256, 0, stream>>>(k_w, kwT, 1024, 1024);
  conv_transpose_kernel<<<1024, 256, 0, stream>>>(v_w, vwT, 1024, 1024);
  conv_transpose_kernel<<<1024, 256, 0, stream>>>(o_w, owT, 1024, 1024);

  // 2) qkv = x @ c_attn_w + b       [8192,3072] f16
  gemm_wmma_kernel<0><<<dim3(24, 64), 256, 0, stream>>>(
      xh, 1024, wAttnT, c_attn_b, qkvh, 3072, 1024, 1.0f);

  // 3) q/k/v projections, scattered straight to [B,H,T,D]; q scaled D^-1/2
  gemm_wmma_kernel<1><<<dim3(8, 64), 256, 0, stream>>>(
      qkvh + 0, 3072, qwT, q_b, qbuf, 0, 1024, 0.125f);
  gemm_wmma_kernel<1><<<dim3(8, 64), 256, 0, stream>>>(
      qkvh + 1024, 3072, kwT, k_b, kbuf, 0, 1024, 1.0f);
  gemm_wmma_kernel<1><<<dim3(8, 64), 256, 0, stream>>>(
      qkvh + 2048, 3072, vwT, v_b, vbuf, 0, 1024, 1.0f);

  // 4) lse init (-1e30 => zero combine weight for uncovered positions).
  //    L0/L1/L2 are contiguous in the workspace carve-out.
  fill_f32_kernel<<<512, 256, 0, stream>>>(L0, -1e30f, 3ll * 131072);

  // 5) dilated attention branches (w, r, nseg): (512,1,8) (1024,2,4) (2048,4,2)
  const size_t attn_lds = (2 * 512 * 64 + 16 * 32 * 32) * sizeof(half_t); // 160 KB
  attn_branch_kernel<<<256, 512, attn_lds, stream>>>(qbuf, kbuf, vbuf, O0, L0, 512, 1, 8);
  attn_branch_kernel<<<128, 512, attn_lds, stream>>>(qbuf, kbuf, vbuf, O1, L1, 1024, 2, 4);
  attn_branch_kernel<<<64, 512, attn_lds, stream>>>(qbuf, kbuf, vbuf, O2, L2, 2048, 4, 2);

  // 6) softmax-weighted merge -> y [B,T,C] f16
  combine_kernel<<<512, 256, 0, stream>>>(O0, O1, O2, L0, L1, L2, yh);

  // 7) out = y @ o_w + o_b   (f32)
  gemm_wmma_kernel<2><<<dim3(8, 64), 256, 0, stream>>>(
      yh, 1024, owT, o_b, (void*)out, 1024, 1024, 1.0f);
}
